// VisibleFusion_75445395522213
// MI455X (gfx1250) — compile-verified, matrix operand order checked by reading, not run on hardware
//
#include <hip/hip_runtime.h>
#include <hip/hip_bf16.h>
#include <math.h>

typedef __attribute__((ext_vector_type(16))) _Float16 v16h;
typedef __attribute__((ext_vector_type(8)))  _Float16 v8h;
typedef __attribute__((ext_vector_type(8)))  float    v8f;

#define NVIEW 9
#define NCH   24
#define HFEAT 120
#define WFEAT 160
#define SIMDIM 72
#define VPW   16              // voxels per wave
#define WAVES 2               // waves per block
#define BLK   (WAVES * 32)

// 16-bit A-fragment (16x32), CDNA5 ISA 7.12.2: per lane-half, elements 0..7
// hold a contiguous K run, elements 8..15 a second contiguous run:
//   lanes 0-15 : K 0..7   and K 16..23   -> 16B chunks 0 and 2 of the row
//   lanes 16-31: K 8..15  and K 24..31   -> 16B chunks 1 and 3 of the row

__global__ void __launch_bounds__(BLK)
vf_main(const int* __restrict__ coords, const float* __restrict__ feats,
        const float* __restrict__ proj_feat, const float* __restrict__ proj_img,
        const float* __restrict__ origin, const float* __restrict__ W_vs,
        const float* __restrict__ b_vs, const float* __restrict__ p_vsz,
        const int* __restrict__ p_scale, const int* __restrict__ p_imh,
        const int* __restrict__ p_imw,
        float* __restrict__ out_fv, float* __restrict__ out_vw,
        float* __restrict__ out_cnt, int N)
{
  __shared__ float    s_feat[WAVES][VPW][NVIEW][NCH];   // masked features (f32)
  __shared__ _Float16 s_fnh [WAVES][VPW][16][32];       // normalized fn, padded
  __shared__ _Float16 s_vish[WAVES][VPW][96];           // vis, padded halves
  __shared__ int4     s_xy  [WAVES][VPW][NVIEW];        // {x0, x1, y0*W, y1*W}
  __shared__ float2   s_ww  [WAVES][VPW][NVIEW];        // {wx, wy}
  __shared__ float    s_mask[WAVES][VPW][NVIEW];
  __shared__ float    s_z   [WAVES][VPW][NVIEW];
  __shared__ float    s_log [WAVES][VPW][16];
  __shared__ float    s_vw  [WAVES][VPW][NVIEW];
  __shared__ float    s_dump[WAVES][2];
  __shared__ _Float16 s_wpadh[16][96];                  // W_vs^T, padded halves
  __shared__ float    s_bpad[16];                       // b_vs, padded

  const int tid  = threadIdx.x;
  const int wid  = tid >> 5;
  const int lane = tid & 31;
  const int m    = lane & 15;        // fragment row (A: M, B: N, C/D: N)
  const int hi   = lane >> 4;        // lane half
  const int c0   = hi;               // 16B chunk index of first K run
  const int c1   = hi + 2;           // 16B chunk index of second K run
  const int n0   = (blockIdx.x * WAVES + wid) * VPW;

  const float vsz  = *p_vsz;
  const float cell = vsz * exp2f((float)(*p_scale));
  const float imw1 = (float)(*p_imw - 1);
  const float imh1 = (float)(*p_imh - 1);
  const float ox0 = origin[0], oy0 = origin[1], oz0 = origin[2];

  // ---- Stage 0: zero padded half buffers; stage W_vs / b_vs into LDS ----
  {
    v8h z8 = {};
    v8h* p = (v8h*)&s_fnh[wid][0][0][0];                // 1024 chunks / wave
    for (int i = lane; i < VPW * 16 * 32 / 8; i += 32) p[i] = z8;
    v8h* q = (v8h*)&s_vish[wid][0][0];                  // 192 chunks / wave
    for (int i = lane; i < VPW * 96 / 8; i += 32) q[i] = z8;
  }
  for (int i = tid; i < 16 * 96; i += BLK) {
    int n = i / 96, k = i - n * 96;
    float val = (k < SIMDIM && n < NVIEW) ? W_vs[k * NVIEW + n] : 0.0f;
    s_wpadh[n][k] = (_Float16)val;
  }
  if (tid < 16) s_bpad[tid] = (tid < NVIEW) ? b_vs[tid] : 0.0f;

  // ---- Stage 1: projections + corner masks, 16 voxels x 9 views ----
  for (int t = lane; t < VPW * NVIEW; t += 32) {
    int vx = t / NVIEW, v = t % NVIEW;
    int n  = n0 + vx;
    int nc = n < N ? n : (N - 1);
    float wx_ = (float)coords[nc * 4 + 1] * vsz + ox0;
    float wy_ = (float)coords[nc * 4 + 2] * vsz + oy0;
    float wz_ = (float)coords[nc * 4 + 3] * vsz + oz0;

    const float* Pi = proj_img + v * 16;
    bool any = false;
#pragma unroll
    for (int cr = 0; cr < 8; ++cr) {
      float px = wx_ + ((cr & 4) ? -0.5f : 0.5f) * cell;
      float py = wy_ + ((cr & 2) ? -0.5f : 0.5f) * cell;
      float pz = wz_ + ((cr & 1) ? -0.5f : 0.5f) * cell;
      float X = Pi[0] * px + Pi[1] * py + Pi[2]  * pz + Pi[3];
      float Y = Pi[4] * px + Pi[5] * py + Pi[6]  * pz + Pi[7];
      float Z = Pi[8] * px + Pi[9] * py + Pi[10] * pz + Pi[11];
      float gx = 2.0f * (X / Z) / imw1 - 1.0f;
      float gy = 2.0f * (Y / Z) / imh1 - 1.0f;
      any = any || (fabsf(gx) <= 1.1f && fabsf(gy) <= 1.1f && Z > 0.0f);
    }
    float mk = any ? 1.0f : 0.0f;

    const float* Pf = proj_feat + v * 16;
    float X = Pf[0] * wx_ + Pf[1] * wy_ + Pf[2]  * wz_ + Pf[3];
    float Y = Pf[4] * wx_ + Pf[5] * wy_ + Pf[6]  * wz_ + Pf[7];
    float Z = Pf[8] * wx_ + Pf[9] * wy_ + Pf[10] * wz_ + Pf[11];
    float fx = fminf(fmaxf(X / Z, 0.0f), (float)(WFEAT - 1));
    float fy = fminf(fmaxf(Y / Z, 0.0f), (float)(HFEAT - 1));
    float fx0 = floorf(fx), fy0 = floorf(fy);
    int x0 = (int)fx0, y0 = (int)fy0;
    int x1 = min(x0 + 1, WFEAT - 1);
    int y1 = min(y0 + 1, HFEAT - 1);
    s_xy[wid][vx][v] = make_int4(x0, x1, y0 * WFEAT, y1 * WFEAT);
    s_ww[wid][vx][v] = make_float2(fx - fx0, fy - fy0);
    s_mask[wid][vx][v] = mk;
    s_z[wid][vx][v] = Z;
  }
  __syncthreads();

  // ---- Stage 2: bilinear gather (feats is L2-resident) ----
  for (int t = lane; t < VPW * NVIEW * NCH; t += 32) {
    int vx  = t / (NVIEW * NCH);
    int rem = t - vx * (NVIEW * NCH);
    int v   = rem / NCH, ch = rem - v * NCH;
    const float* fm = feats + ((size_t)v * NCH + ch) * (HFEAT * WFEAT);
    int4   xy = s_xy[wid][vx][v];
    float2 ww = s_ww[wid][vx][v];
    float f00 = fm[xy.z + xy.x], f01 = fm[xy.z + xy.y];
    float f10 = fm[xy.w + xy.x], f11 = fm[xy.w + xy.y];
    float f = f00 * (1.0f - ww.x) * (1.0f - ww.y) + f01 * ww.x * (1.0f - ww.y)
            + f10 * (1.0f - ww.x) * ww.y          + f11 * ww.x * ww.y;
    s_feat[wid][vx][v][ch] = f * s_mask[wid][vx][v];
  }
  __syncthreads();

  // ---- Stage 3: normalize, write fn as padded halves (fragment-ready) ----
  for (int t = lane; t < VPW * NVIEW; t += 32) {
    int vx = t / NVIEW, v = t % NVIEW;
    float ss = 0.0f;
    float fr[NCH];
#pragma unroll
    for (int ch = 0; ch < NCH; ++ch) {
      fr[ch] = s_feat[wid][vx][v][ch];
      ss += fr[ch] * fr[ch];
    }
    float inv = 1.0f / (sqrtf(ss) + 1e-10f);
    v8h h8[3];
#pragma unroll
    for (int c = 0; c < 3; ++c)
#pragma unroll
      for (int e = 0; e < 8; ++e)
        h8[c][e] = (_Float16)(fr[c * 8 + e] * inv);
    v8h* row = (v8h*)&s_fnh[wid][vx][v][0];
    row[0] = h8[0]; row[1] = h8[1]; row[2] = h8[2];   // K 24..31 stay zero
  }
  __syncthreads();

  // ---- Stage 4: sim = fn * fn^T per voxel via WMMA (A == B fragment) ----
#pragma unroll 1
  for (int vx = 0; vx < VPW; ++vx) {
    const v8h* row = (const v8h*)&s_fnh[wid][vx][m][0];
    v8h lo8 = row[c0];
    v8h hi8 = row[c1];
    v16h a = __builtin_shufflevector(lo8, hi8, 0, 1, 2, 3, 4, 5, 6, 7,
                                     8, 9, 10, 11, 12, 13, 14, 15);
    v8f c = {};
    c = __builtin_amdgcn_wmma_f32_16x16x32_f16(false, a, false, a,
                                               (short)0, c, false, false);
    // branchless scatter: lane holds N=w=m, VGPR r holds M=v=r+8*hi
    int w = m;
#pragma unroll
    for (int r = 0; r < 8; ++r) {
      int v = r + 8 * hi;
      int f = v * NVIEW + w;
      int aa = f / 10, bb = f - aa * 10;
      int idx = aa * 9 + bb - 1;
      bool valid = (w < NVIEW) & (v < NVIEW) & (v != w);
      _Float16* dst = valid ? &s_vish[wid][vx][idx]
                            : (_Float16*)&s_dump[wid][0];
      *dst = (_Float16)c[r];
    }
  }
  __syncthreads();

  // ---- Stage 5: logits = vis @ W_vs + b, batched over 16 voxels ----
  v16h bw[3];
  {
    const v8h* wrow = (const v8h*)&s_wpadh[m][0];
#pragma unroll
    for (int kb = 0; kb < 3; ++kb) {
      v8h lo8 = wrow[kb * 4 + c0];
      v8h hi8 = wrow[kb * 4 + c1];
      bw[kb] = __builtin_shufflevector(lo8, hi8, 0, 1, 2, 3, 4, 5, 6, 7,
                                       8, 9, 10, 11, 12, 13, 14, 15);
    }
  }
  v8f cl;
  {
    float bv = s_bpad[m];              // C/D layout: N = lane&15
#pragma unroll
    for (int r = 0; r < 8; ++r) cl[r] = bv;
  }
  {
    const v8h* vrow = (const v8h*)&s_vish[wid][m][0];
#pragma unroll
    for (int kb = 0; kb < 3; ++kb) {
      v8h lo8 = vrow[kb * 4 + c0];
      v8h hi8 = vrow[kb * 4 + c1];
      v16h av = __builtin_shufflevector(lo8, hi8, 0, 1, 2, 3, 4, 5, 6, 7,
                                        8, 9, 10, 11, 12, 13, 14, 15);
      cl = __builtin_amdgcn_wmma_f32_16x16x32_f16(false, av, false, bw[kb],
                                                  (short)0, cl, false, false);
    }
  }
#pragma unroll
  for (int r = 0; r < 8; ++r)
    s_log[wid][r + 8 * hi][m] = cl[r];
  __syncthreads();

  // ---- Stage 6: softmax * mask, zmean, count ----
  if (lane < VPW) {
    int vx = lane;
    int n  = n0 + vx;
    float mx = -3.0e38f;
#pragma unroll
    for (int w = 0; w < NVIEW; ++w) mx = fmaxf(mx, s_log[wid][vx][w]);
    float ssum = 0.0f;
    float e9[NVIEW];
#pragma unroll
    for (int w = 0; w < NVIEW; ++w) {
      e9[w] = __expf(s_log[wid][vx][w] - mx);
      ssum += e9[w];
    }
    float inv = 1.0f / ssum;
    float smsk = 0.0f, sz = 0.0f;
#pragma unroll
    for (int w = 0; w < NVIEW; ++w) {
      float vw = e9[w] * inv * s_mask[wid][vx][w];
      s_vw[wid][vx][w] = vw;
      if (n < N) out_vw[(size_t)n * NVIEW + w] = vw;
      smsk += s_mask[wid][vx][w];
      sz   += s_z[wid][vx][w] * s_mask[wid][vx][w];
    }
    if (n < N) {
      out_cnt[n] = smsk;
      out_fv[(size_t)n * 25 + 24] = sz / (smsk + 1e-10f);  // zmean staged here
    }
  }
  __syncthreads();

  // ---- Stage 7: fused features ----
  for (int t = lane; t < VPW * NCH; t += 32) {
    int vx = t / NCH, ch = t - vx * NCH;
    int n  = n0 + vx;
    float s = 0.0f;
#pragma unroll
    for (int w = 0; w < NVIEW; ++w)
      s += s_feat[wid][vx][w][ch] * s_vw[wid][vx][w];
    if (n < N) out_fv[(size_t)n * 25 + ch] = s;
  }
}

// ---- deterministic two-stage reduction of zmean stats ----
__global__ void __launch_bounds__(256)
vf_reduce1(const float* __restrict__ fv, int N, float* __restrict__ part) {
  __shared__ float sh[3][256];
  float s1 = 0.0f, s2 = 0.0f, c = 0.0f;
  for (int i = blockIdx.x * blockDim.x + threadIdx.x; i < N;
       i += gridDim.x * blockDim.x) {
    float z = fv[(size_t)i * 25 + 24];
    if (z > 0.0f) { s1 += z; s2 += z * z; c += 1.0f; }
  }
  int t = threadIdx.x;
  sh[0][t] = s1; sh[1][t] = s2; sh[2][t] = c;
  __syncthreads();
  for (int off = 128; off > 0; off >>= 1) {
    if (t < off) {
      sh[0][t] += sh[0][t + off];
      sh[1][t] += sh[1][t + off];
      sh[2][t] += sh[2][t + off];
    }
    __syncthreads();
  }
  if (t == 0) {
    part[blockIdx.x * 3 + 0] = sh[0][0];
    part[blockIdx.x * 3 + 1] = sh[1][0];
    part[blockIdx.x * 3 + 2] = sh[2][0];
  }
}

__global__ void __launch_bounds__(256)
vf_reduce2(float* __restrict__ part) {
  __shared__ float sh[3][256];
  int t = threadIdx.x;
  sh[0][t] = part[t * 3 + 0];
  sh[1][t] = part[t * 3 + 1];
  sh[2][t] = part[t * 3 + 2];
  __syncthreads();
  for (int off = 128; off > 0; off >>= 1) {
    if (t < off) {
      sh[0][t] += sh[0][t + off];
      sh[1][t] += sh[1][t + off];
      sh[2][t] += sh[2][t + off];
    }
    __syncthreads();
  }
  if (t == 0) {
    float S1 = sh[0][0], S2 = sh[1][0], Cn = sh[2][0];
    float mu = S1 / fmaxf(Cn, 1.0f);
    float var = S2 - 2.0f * mu * S1 + Cn * mu * mu;
    float sd = sqrtf(fmaxf(var, 0.0f)) + 1e-5f;
    part[768] = mu;
    part[769] = sd;
  }
}

__global__ void __launch_bounds__(256)
vf_znorm(float* __restrict__ fv, int N, const float* __restrict__ musd) {
  int i = blockIdx.x * blockDim.x + threadIdx.x;
  if (i >= N) return;
  float mu = musd[0], sd = musd[1];
  float z = fv[(size_t)i * 25 + 24];
  fv[(size_t)i * 25 + 24] = (z > 0.0f) ? (z - mu) / sd : 0.0f;
}

extern "C" void kernel_launch(void* const* d_in, const int* in_sizes, int n_in,
                              void* d_out, int out_size, void* d_ws, size_t ws_size,
                              hipStream_t stream) {
  const int*   coords    = (const int*)  d_in[0];
  const float* feats     = (const float*)d_in[1];
  const float* proj_feat = (const float*)d_in[2];
  const float* proj_img  = (const float*)d_in[3];
  const float* origin    = (const float*)d_in[4];
  const float* W_vs      = (const float*)d_in[5];
  const float* b_vs      = (const float*)d_in[6];
  const float* vsz       = (const float*)d_in[7];
  const int*   scale     = (const int*)  d_in[8];
  const int*   imh       = (const int*)  d_in[9];
  const int*   imw       = (const int*)  d_in[10];

  int N = in_sizes[0] / 4;
  float* out     = (float*)d_out;
  float* out_fv  = out;                      // (N, 25)
  float* out_vw  = out + (size_t)N * 25;     // (N, 9)
  float* out_cnt = out + (size_t)N * 34;     // (N,)
  float* ws      = (float*)d_ws;             // 770 floats used

  int blocks1 = (N + WAVES * VPW - 1) / (WAVES * VPW);
  vf_main<<<blocks1, BLK, 0, stream>>>(coords, feats, proj_feat, proj_img,
                                       origin, W_vs, b_vs, vsz, scale, imh, imw,
                                       out_fv, out_vw, out_cnt, N);
  vf_reduce1<<<256, 256, 0, stream>>>(out_fv, N, ws);
  vf_reduce2<<<1, 256, 0, stream>>>(ws);
  vf_znorm<<<(N + 255) / 256, 256, 0, stream>>>(out_fv, N, ws + 768);
}
